// SpatialGraphAttentionV3_63952063037504
// MI455X (gfx1250) — matrix operand order repaired and checked by reading, not compiled, not run on hardware
//
#include <hip/hip_runtime.h>
#include <hip/hip_bf16.h>
#include <math.h>

// ---------------------------------------------------------------------------
// SpatialGraphAttentionV3 for MI455X (gfx1250, wave32, WMMA bf16 path)
// Async Global->LDS staging (ASYNCcnt) + double-buffered WMMA pipeline.
// ---------------------------------------------------------------------------
#define D_MODEL 896
#define N_HEADS 14
#define D_HEAD  64
#define BB      8
#define NN      1024

typedef __bf16 bf16;
typedef __attribute__((ext_vector_type(16))) __bf16 v16bf;
typedef __attribute__((ext_vector_type(8)))  __bf16 bf16x8;
typedef __attribute__((ext_vector_type(8)))  float  v8f;

__device__ __forceinline__ bf16 f2bf(float f) {
  unsigned u = __builtin_bit_cast(unsigned, f);
  unsigned r = u + 0x7FFFu + ((u >> 16) & 1u);   // round-to-nearest-even
  unsigned short h = (unsigned short)(r >> 16);
  return __builtin_bit_cast(bf16, h);
}

__device__ __forceinline__ v8f wmma_bf(v16bf a, v16bf b, v8f c) {
  return __builtin_amdgcn_wmma_f32_16x16x32_bf16(false, a, false, b, (short)0, c,
                                                 false, false);
}

// 32-bit LDS offset of a generic pointer into shared memory (ISA: addr[31:0]).
__device__ __forceinline__ unsigned lds_lo(const void* p) {
  return (unsigned)(unsigned long long)p;
}

// Async DMA of 16 bytes global -> LDS, tracked by ASYNCcnt (per-lane address).
__device__ __forceinline__ void async_b128(unsigned lds_off, const bf16* g) {
  asm volatile("global_load_async_to_lds_b128 %0, %1, off"
               :: "v"(lds_off), "v"(g) : "memory");
}

// A fragment: 16(M) x 32(K) bf16, src row-major [m][k], row stride ld elements.
__device__ __forceinline__ v16bf load_afrag(const bf16* __restrict__ src, int ld,
                                            int k0, int lane) {
  int m = lane & 15, g = lane >> 4;
  const bf16* p0 = src + (size_t)m * ld + k0 + g * 16;
  const bf16* p1 = src + (size_t)m * ld + k0 + g * 16 + 8;
  bf16x8 lo = *(const bf16x8*)p0;
  bf16x8 hi = *(const bf16x8*)p1;
  v16bf a;
#pragma unroll
  for (int e = 0; e < 8; ++e) { a[e] = lo[e]; a[e + 8] = hi[e]; }
  return a;
}

// B fragment: 32(K) x 16(N) bf16 from k-major source [n][k], stride ld.
__device__ __forceinline__ v16bf load_bfrag(const bf16* __restrict__ src, int ld,
                                            int n0, int k0, int lane) {
  int n = lane & 15, hh = lane >> 4;
  return *(const v16bf*)(src + (size_t)(n0 + n) * ld + k0 + 16 * hh);
}

// ---------------------------------------------------------------------------
// Kernel 1: LayerNorm -> bf16
// ---------------------------------------------------------------------------
__global__ __launch_bounds__(256) void ln_kernel(const float* __restrict__ x,
                                                 const float* __restrict__ g,
                                                 const float* __restrict__ bta,
                                                 bf16* __restrict__ xn) {
  int row = blockIdx.x;
  int tid = threadIdx.x;
  const float* xr = x + (size_t)row * D_MODEL;
  float s = 0.f, s2 = 0.f;
  for (int i = tid; i < D_MODEL; i += 256) {
    float v = xr[i];
    s += v; s2 += v * v;
  }
  __shared__ float r1[256], r2[256];
  r1[tid] = s; r2[tid] = s2;
  __syncthreads();
  for (int st = 128; st > 0; st >>= 1) {
    if (tid < st) { r1[tid] += r1[tid + st]; r2[tid] += r2[tid + st]; }
    __syncthreads();
  }
  float mu  = r1[0] * (1.0f / D_MODEL);
  float var = r2[0] * (1.0f / D_MODEL) - mu * mu;
  float rs  = rsqrtf(var + 1e-5f);
  bf16* xo = xn + (size_t)row * D_MODEL;
  for (int i = tid; i < D_MODEL; i += 256)
    xo[i] = f2bf((xr[i] - mu) * rs * g[i] + bta[i]);
}

// ---------------------------------------------------------------------------
// Kernel 2: weight convert + transpose (f32 [k][n] -> bf16 [n][k])
// ---------------------------------------------------------------------------
__global__ __launch_bounds__(256) void wconv_kernel(const float* __restrict__ W,
                                                    bf16* __restrict__ Wt) {
  int idx = blockIdx.x * 256 + threadIdx.x;
  if (idx < D_MODEL * D_MODEL) {
    int k = idx / D_MODEL, n = idx % D_MODEL;
    Wt[(size_t)n * D_MODEL + k] = f2bf(W[(size_t)k * D_MODEL + n]);
  }
}

// ---------------------------------------------------------------------------
// Kernel 3: edge trig precompute.
// reverse_bearings = (bearings.T + 180) % 360 -> cos = -cos(bear), sin = -sin.
// ---------------------------------------------------------------------------
__global__ __launch_bounds__(256) void edge_kernel(const float* __restrict__ bear,
                                                   float* __restrict__ crb,
                                                   float* __restrict__ srb) {
  int idx = blockIdx.x * 256 + threadIdx.x;
  if (idx < NN * NN) {
    int q = idx / NN, k = idx % NN;
    float a = bear[(size_t)k * NN + q] * 0.017453292519943295f;
    crb[idx] = -__cosf(a);
    srb[idx] = -__sinf(a);
  }
}

__global__ __launch_bounds__(256) void wind_kernel(const float* __restrict__ wd,
                                                   float* __restrict__ cwd,
                                                   float* __restrict__ swd) {
  int idx = blockIdx.x * 256 + threadIdx.x;
  if (idx < BB * NN) {
    float a = wd[idx] * 0.017453292519943295f;
    cwd[idx] = __cosf(a);
    swd[idx] = __sinf(a);
  }
}

// ---------------------------------------------------------------------------
// Kernel 4: tiled WMMA GEMM, 128x64 macro tile (one head per N tile).
// Weight panel (64 n x 32 k) staged to LDS once per WG via async DMA,
// double-buffered; A fragments software-pipelined in registers.
// mode 0: Q (rope, *1/8, -> [b][h][n][d])   mode 1: K (rope, -> [b][h][n][d])
// mode 2: V (-> d-major [b][h][d][n])       mode 3: out proj (+residual, f32)
// ---------------------------------------------------------------------------
#define PROJ_NK (D_MODEL / 32)   // 28 k-steps

__global__ __launch_bounds__(256) void proj_kernel(const bf16* __restrict__ A,
                                                   const bf16* __restrict__ Bt,
                                                   bf16* __restrict__ outh,
                                                   float* __restrict__ outf,
                                                   const float* __restrict__ resid,
                                                   int mode) {
  __shared__ __align__(16) bf16 ldsB[2][64 * 32];   // [buf][n][k] panel, 4KB each
  int nb = blockIdx.x;                 // 0..13 (head)
  int mb = blockIdx.y;                 // 0..63
  int tid = threadIdx.x;
  int wave = tid >> 5, lane = tid & 31;
  int nloc = lane & 15, hv = lane >> 4;
  int m0 = mb * 128 + wave * 16;
  int n0 = nb * 64;

  // panel copy assignment: 64 rows x 4 chunks of 16B = 256 lanes
  int prow = tid >> 2, pchk = tid & 3;
  const bf16* psrc0 = Bt + (size_t)(n0 + prow) * D_MODEL + pchk * 8;
  unsigned pdst0 = lds_lo(&ldsB[0][prow * 32 + pchk * 8]);
  unsigned pdst1 = lds_lo(&ldsB[1][prow * 32 + pchk * 8]);

  v8f acc[4];
#pragma unroll
  for (int j = 0; j < 4; ++j)
#pragma unroll
    for (int e = 0; e < 8; ++e) acc[j][e] = 0.f;

  const bf16* Arow = A + (size_t)m0 * D_MODEL;

  async_b128(pdst0, psrc0);                       // prologue: panel for k=0
  v16bf va = load_afrag(Arow, D_MODEL, 0, lane);  // A frag for k=0

  for (int kk = 0; kk < PROJ_NK; ++kk) {
    int cur = kk & 1;
    if (kk + 1 < PROJ_NK) {
      // issue next panel, then wait for the current one (in-order completion)
      async_b128(cur ? pdst0 : pdst1, psrc0 + (kk + 1) * 32);
      asm volatile("s_wait_asynccnt 0x1" ::: "memory");
    } else {
      asm volatile("s_wait_asynccnt 0x0" ::: "memory");
    }
    __syncthreads();

    v16bf va_n;
    if (kk + 1 < PROJ_NK) {
      va_n = load_afrag(Arow, D_MODEL, (kk + 1) * 32, lane);  // overlap w/ WMMA
      __builtin_prefetch(Arow + (size_t)(lane & 15) * D_MODEL + (kk + 2) * 32, 0, 3);
    }
    const bf16* panel = ldsB[cur];
#pragma unroll
    for (int j = 0; j < 4; ++j) {
      v16bf vb = load_bfrag(panel, 32, j * 16, 0, lane);
      acc[j] = wmma_bf(va, vb, acc[j]);
    }
    va = va_n;
    __syncthreads();   // protect panel buffer before it is re-filled
  }

  if (mode <= 1) {
    // RoPE: pair (d, d+32) lives in fragments (j, j+2) at the same lane/row.
    float scale = (mode == 0) ? 0.125f : 1.0f;   // fold 1/sqrt(D_HEAD) into Q
#pragma unroll
    for (int j = 0; j < 2; ++j) {
      int i = j * 16 + nloc;                     // rotary index 0..31
      float inv = __expf(-(float)i * (9.210340371976184f / 32.0f)); // base^{-i/32}
#pragma unroll
      for (int r = 0; r < 8; ++r) {
        int m = m0 + r + 8 * hv;
        int pos = m & (NN - 1);
        float th = (float)pos * inv;
        float c = __cosf(th), s = __sinf(th);
        float x1 = acc[j][r], x2 = acc[j + 2][r];
        acc[j][r]     = (x1 * c - x2 * s) * scale;
        acc[j + 2][r] = (x2 * c + x1 * s) * scale;
      }
    }
#pragma unroll
    for (int j = 0; j < 4; ++j)
#pragma unroll
      for (int r = 0; r < 8; ++r) {
        int m = m0 + r + 8 * hv;
        int b = m >> 10, pos = m & (NN - 1);
        int d = j * 16 + nloc;
        outh[(((size_t)b * N_HEADS + nb) * NN + pos) * D_HEAD + d] = f2bf(acc[j][r]);
      }
  } else if (mode == 2) {
#pragma unroll
    for (int j = 0; j < 4; ++j)
#pragma unroll
      for (int r = 0; r < 8; ++r) {
        int m = m0 + r + 8 * hv;
        int b = m >> 10, pos = m & (NN - 1);
        int d = j * 16 + nloc;
        outh[(((size_t)b * N_HEADS + nb) * D_HEAD + d) * NN + pos] = f2bf(acc[j][r]);
      }
  } else {
#pragma unroll
    for (int j = 0; j < 4; ++j)
#pragma unroll
      for (int r = 0; r < 8; ++r) {
        int m = m0 + r + 8 * hv;
        size_t idx = (size_t)m * D_MODEL + n0 + j * 16 + nloc;
        outf[idx] = acc[j][r] + resid[idx];
      }
  }
}

// ---------------------------------------------------------------------------
// Kernel 5: flash attention with mask + wind bias.
// Grid: (N/128, H, B); 8 waves x 16 q-rows; 32-wide kv steps.
// K (32x64) and V^T (64x32) tiles async-DMA'd to LDS, double-buffered.
// ---------------------------------------------------------------------------
__global__ __launch_bounds__(256) void attn_kernel(const bf16* __restrict__ Q,
                                                   const bf16* __restrict__ K,
                                                   const bf16* __restrict__ VT,
                                                   const float* __restrict__ adj,
                                                   const float* __restrict__ crb,
                                                   const float* __restrict__ srb,
                                                   const float* __restrict__ cwd,
                                                   const float* __restrict__ swd,
                                                   const float* __restrict__ ww,
                                                   const float* __restrict__ wbb,
                                                   bf16* __restrict__ outA) {
  __shared__ __align__(16) bf16 ldsK[2][32 * 64];   // [buf][kv][d]
  __shared__ __align__(16) bf16 ldsV[2][64 * 32];   // [buf][d][kv]
  __shared__ __align__(16) bf16 ldsP[8][16 * 32];   // per-wave P transpose
  int qb = blockIdx.x, h = blockIdx.y, b = blockIdx.z;
  int tid = threadIdx.x, wave = tid >> 5, lane = tid & 31;
  int nloc = lane & 15, hv = lane >> 4;
  int q0 = qb * 128 + wave * 16;

  const bf16* Qh = Q  + (((size_t)b * N_HEADS + h) * NN) * D_HEAD;
  const bf16* Kh = K  + (((size_t)b * N_HEADS + h) * NN) * D_HEAD;
  const bf16* Vh = VT + (((size_t)b * N_HEADS + h) * D_HEAD) * NN;

  // async copy assignment
  int krow = tid >> 3, kchk = tid & 7;     // 32 rows x 8 chunks (K tile)
  int vrow = tid >> 2, vchk = tid & 3;     // 64 rows x 4 chunks (V tile)
  const bf16* ksrc = Kh + (size_t)krow * D_HEAD + kchk * 8;
  const bf16* vsrc = Vh + (size_t)vrow * NN + vchk * 8;
  unsigned kdst[2], vdst[2];
  kdst[0] = lds_lo(&ldsK[0][krow * 64 + kchk * 8]);
  kdst[1] = lds_lo(&ldsK[1][krow * 64 + kchk * 8]);
  vdst[0] = lds_lo(&ldsV[0][vrow * 32 + vchk * 8]);
  vdst[1] = lds_lo(&ldsV[1][vrow * 32 + vchk * 8]);

  v16bf qa0 = load_afrag(Qh + (size_t)q0 * D_HEAD, D_HEAD, 0, lane);
  v16bf qa1 = load_afrag(Qh + (size_t)q0 * D_HEAD, D_HEAD, 32, lane);

  float w0 = ww[h], w1 = ww[N_HEADS + h], wb0 = wbb[h];

  v8f o[4];
  float mrun[8], lrun[8];
#pragma unroll
  for (int j = 0; j < 4; ++j)
#pragma unroll
    for (int e = 0; e < 8; ++e) o[j][e] = 0.f;
#pragma unroll
  for (int r = 0; r < 8; ++r) { mrun[r] = -1e30f; lrun[r] = 0.f; }

  bf16* lp = ldsP[wave];

  // prologue: stage kv=0 tiles
  async_b128(kdst[0], ksrc);
  async_b128(vdst[0], vsrc);

  for (int step = 0; step < NN / 32; ++step) {
    int kv = step * 32, cur = step & 1;
    if (step + 1 < NN / 32) {
      async_b128(kdst[cur ^ 1], ksrc + (size_t)(kv + 32) * D_HEAD);
      async_b128(vdst[cur ^ 1], vsrc + kv + 32);
      asm volatile("s_wait_asynccnt 0x2" ::: "memory");
    } else {
      asm volatile("s_wait_asynccnt 0x0" ::: "memory");
    }
    __syncthreads();

    const bf16* Kt = ldsK[cur];
    const bf16* Vt = ldsV[cur];
    v16bf kb00 = load_bfrag(Kt, D_HEAD, 0,  0,  lane);
    v16bf kb01 = load_bfrag(Kt, D_HEAD, 0,  32, lane);
    v16bf kb10 = load_bfrag(Kt, D_HEAD, 16, 0,  lane);
    v16bf kb11 = load_bfrag(Kt, D_HEAD, 16, 32, lane);
    v8f s0, s1;
#pragma unroll
    for (int e = 0; e < 8; ++e) { s0[e] = 0.f; s1[e] = 0.f; }
    s0 = wmma_bf(qa0, kb00, s0); s0 = wmma_bf(qa1, kb01, s0);
    s1 = wmma_bf(qa0, kb10, s1); s1 = wmma_bf(qa1, kb11, s1);

    float fcor[8];
#pragma unroll
    for (int r = 0; r < 8; ++r) {
      int q = q0 + r + 8 * hv;
      size_t rowo = (size_t)q * NN;
      int k0i = kv + nloc, k1i = kv + 16 + nloc;
      float cq = cwd[b * NN + q], sq = swd[b * NN + q];
      float a0 = adj[rowo + k0i], a1 = adj[rowo + k1i];
      float al0 = cq * crb[rowo + k0i] + sq * srb[rowo + k0i];
      float al1 = cq * crb[rowo + k1i] + sq * srb[rowo + k1i];
      float b0 = tanhf(al0 * w0 + a0 * w1 + wb0);
      float b1 = tanhf(al1 * w0 + a1 * w1 + wb0);
      float v0 = (a0 > 0.f ? s0[r] : -1e30f) + b0;
      float v1 = (a1 > 0.f ? s1[r] : -1e30f) + b1;
      float tm = fmaxf(v0, v1);
#pragma unroll
      for (int off = 1; off < 16; off <<= 1)
        tm = fmaxf(tm, __shfl_xor(tm, off, 32));
      float newm = fmaxf(mrun[r], tm);
      float f  = __expf(mrun[r] - newm);
      float e0 = __expf(v0 - newm), e1 = __expf(v1 - newm);
      float rs = e0 + e1;
#pragma unroll
      for (int off = 1; off < 16; off <<= 1)
        rs += __shfl_xor(rs, off, 32);
      lrun[r] = lrun[r] * f + rs;
      mrun[r] = newm;
      fcor[r] = f;
      int mrow = r + 8 * hv;
      lp[mrow * 32 + nloc]      = f2bf(e0);
      lp[mrow * 32 + 16 + nloc] = f2bf(e1);
    }
#pragma unroll
    for (int j = 0; j < 4; ++j)
#pragma unroll
      for (int r = 0; r < 8; ++r) o[j][r] *= fcor[r];

    v16bf pa = load_afrag(lp, 32, 0, lane);
#pragma unroll
    for (int j = 0; j < 4; ++j) {
      v16bf vb = load_bfrag(Vt, 32, j * 16, 0, lane);
      o[j] = wmma_bf(pa, vb, o[j]);
    }
    __syncthreads();   // protect K/V buffers before refill
  }

#pragma unroll
  for (int r = 0; r < 8; ++r) {
    float inv = 1.0f / lrun[r];
#pragma unroll
    for (int j = 0; j < 4; ++j) o[j][r] *= inv;
  }
#pragma unroll
  for (int j = 0; j < 4; ++j)
#pragma unroll
    for (int r = 0; r < 8; ++r) {
      int q = q0 + r + 8 * hv;
      int col = h * 64 + j * 16 + nloc;
      outA[((size_t)b * NN + q) * D_MODEL + col] = f2bf(o[j][r]);
    }
}

// ---------------------------------------------------------------------------
extern "C" void kernel_launch(void* const* d_in, const int* in_sizes, int n_in,
                              void* d_out, int out_size, void* d_ws, size_t ws_size,
                              hipStream_t stream) {
  (void)in_sizes; (void)n_in; (void)out_size; (void)ws_size;
  const float* node = (const float*)d_in[0];
  const float* adj  = (const float*)d_in[1];
  const float* wind = (const float*)d_in[2];
  const float* bear = (const float*)d_in[3];
  const float* Wq   = (const float*)d_in[4];
  const float* Wk   = (const float*)d_in[5];
  const float* Wv   = (const float*)d_in[6];
  const float* Wo   = (const float*)d_in[7];
  const float* lng  = (const float*)d_in[8];
  const float* lnb  = (const float*)d_in[9];
  const float* ww   = (const float*)d_in[10];
  const float* wb   = (const float*)d_in[11];
  float* out = (float*)d_out;

  char* p = (char*)d_ws;
  auto bump = [&](size_t bytes) {
    char* r = p;
    p += (bytes + 255) & ~(size_t)255;
    return r;
  };
  const size_t tokD = (size_t)BB * NN * D_MODEL;          // 8192*896
  bf16* xn  = (bf16*)bump(tokD * 2);
  bf16* WqT = (bf16*)bump((size_t)D_MODEL * D_MODEL * 2);
  bf16* WkT = (bf16*)bump((size_t)D_MODEL * D_MODEL * 2);
  bf16* WvT = (bf16*)bump((size_t)D_MODEL * D_MODEL * 2);
  bf16* WoT = (bf16*)bump((size_t)D_MODEL * D_MODEL * 2);
  bf16* Qh  = (bf16*)bump(tokD * 2);
  bf16* Kh  = (bf16*)bump(tokD * 2);
  bf16* Vt  = (bf16*)bump(tokD * 2);
  float* crb = (float*)bump((size_t)NN * NN * 4);
  float* srb = (float*)bump((size_t)NN * NN * 4);
  float* cwd = (float*)bump((size_t)BB * NN * 4);
  float* swd = (float*)bump((size_t)BB * NN * 4);
  bf16* Ao = xn;   // xn dead after V projection; reuse for attention output

  ln_kernel<<<BB * NN, 256, 0, stream>>>(node, lng, lnb, xn);
  int wblk = (D_MODEL * D_MODEL + 255) / 256;
  wconv_kernel<<<wblk, 256, 0, stream>>>(Wq, WqT);
  wconv_kernel<<<wblk, 256, 0, stream>>>(Wk, WkT);
  wconv_kernel<<<wblk, 256, 0, stream>>>(Wv, WvT);
  wconv_kernel<<<wblk, 256, 0, stream>>>(Wo, WoT);
  edge_kernel<<<(NN * NN + 255) / 256, 256, 0, stream>>>(bear, crb, srb);
  wind_kernel<<<(BB * NN + 255) / 256, 256, 0, stream>>>(wind, cwd, swd);

  dim3 pg(N_HEADS, (BB * NN) / 128);
  proj_kernel<<<pg, 256, 0, stream>>>(xn, WqT, Qh, nullptr, nullptr, 0);
  proj_kernel<<<pg, 256, 0, stream>>>(xn, WkT, Kh, nullptr, nullptr, 1);
  proj_kernel<<<pg, 256, 0, stream>>>(xn, WvT, Vt, nullptr, nullptr, 2);

  dim3 ag(NN / 128, N_HEADS, BB);
  attn_kernel<<<ag, 256, 0, stream>>>(Qh, Kh, Vt, adj, crb, srb, cwd, swd, ww, wb, Ao);

  proj_kernel<<<pg, 256, 0, stream>>>(Ao, WoT, nullptr, out, node, 3);
}
